// GIN_81527069213092
// MI455X (gfx1250) — matrix operand order repaired and checked by reading, not compiled, not run on hardware
//
#include <hip/hip_runtime.h>

#define NN   16384
#define IND  256
#define HIDD 128
#define OUTD 16

typedef __attribute__((ext_vector_type(2)))  float  v2f;
typedef __attribute__((ext_vector_type(8)))  float  v8f;
typedef __attribute__((ext_vector_type(16))) __bf16 v16bf;
typedef __attribute__((ext_vector_type(2)))  __bf16 v2bf;
typedef __attribute__((__vector_size__(16))) int    vi4;

union AB16 { unsigned u[8]; v16bf v; };

// ---- feature probes (compile-safe: fall back to sync paths on miss) --------
#if defined(__has_builtin)
#  if __has_builtin(__builtin_amdgcn_global_load_async_to_lds_b128) && \
      __has_builtin(__builtin_amdgcn_s_wait_asynccnt)
#    define HAVE_ASYNC 1
#  else
#    define HAVE_ASYNC 0
#  endif
#  if __has_builtin(__builtin_amdgcn_cvt_pk_bf16_f32)
#    define HAVE_CVTPK 1
#  else
#    define HAVE_CVTPK 0
#  endif
#else
#  define HAVE_ASYNC 0
#  define HAVE_CVTPK 0
#endif

#define AS1 __attribute__((address_space(1)))
#define AS3 __attribute__((address_space(3)))

__device__ __forceinline__ void cp_async_b128(void* lds, const void* gsrc) {
#if HAVE_ASYNC
    __builtin_amdgcn_global_load_async_to_lds_b128(
        (AS1 vi4*)gsrc, (AS3 vi4*)lds, 0, 0);
#else
    *(uint4*)lds = *(const uint4*)gsrc;
#endif
}
__device__ __forceinline__ void cp_async_wait() {
#if HAVE_ASYNC
    __builtin_amdgcn_s_wait_asynccnt(0);
#endif
}

__device__ __forceinline__ unsigned pk_bf16(float a, float b) {
#if HAVE_CVTPK
    v2bf r = __builtin_amdgcn_cvt_pk_bf16_f32(a, b);
    return __builtin_bit_cast(unsigned, r);
#else
    unsigned ua = __float_as_uint(a);
    unsigned ub = __float_as_uint(b);
    ua = (ua + 0x7FFFu + ((ua >> 16) & 1u)) >> 16;   // RNE f32 -> bf16
    ub = (ub + 0x7FFFu + ((ub >> 16) & 1u)) >> 16;
    return (ua & 0xFFFFu) | (ub << 16);
#endif
}

// ---------------------------------------------------------------------------
// Exact f32 WMMA GEMM: out[16384][128] = in[16384][K] @ W[K][128] + bias (opt relu)
// 128 threads (4 waves), 64 rows/block. Double-buffered async-to-LDS staging.
// ---------------------------------------------------------------------------
template<int K, bool RELU>
__global__ __launch_bounds__(128)
void gemm_f32_k(const float* __restrict__ in, const float* __restrict__ W,
                const float* __restrict__ bias, float* __restrict__ out)
{
    __shared__ float As[2][64][36];    // pitch 36: conflict-free v2f A-frag reads
    __shared__ float Bs[2][32][136];   // pitch 136: lane-halves hit disjoint banks

    const int tid  = threadIdx.x;
    const int lane = tid & 31;
    const int wv   = tid >> 5;
    const int m0   = blockIdx.x * 64;
    const int half = lane >> 4;
    const int l15  = lane & 15;

    v8f c[8];
    #pragma unroll
    for (int j = 0; j < 8; ++j) c[j] = (v8f)0.0f;

    auto stage = [&](int buf, int kb) {
        #pragma unroll
        for (int i = 0; i < 4; ++i) {                  // A tile 64x32 f32
            int lin = i * 128 + tid;
            int r = lin >> 3, cc = (lin & 7) * 4;
            cp_async_b128(&As[buf][r][cc], in + (size_t)(m0 + r) * K + kb + cc);
        }
        #pragma unroll
        for (int i = 0; i < 8; ++i) {                  // W tile 32x128 f32
            int lin = i * 128 + tid;
            int r = lin >> 5, cc = (lin & 31) * 4;
            cp_async_b128(&Bs[buf][r][cc], W + (size_t)(kb + r) * HIDD + cc);
        }
    };

    stage(0, 0);
    cp_async_wait();
    __syncthreads();

    for (int kb = 0; kb < K; kb += 32) {
        const int cur = (kb >> 5) & 1, nxt = cur ^ 1;
        const bool more = (kb + 32 < K);
        if (more) stage(nxt, kb + 32);                 // overlap with WMMAs below

        #pragma unroll
        for (int ks = 0; ks < 8; ++ks) {
            const int ka = ks * 4 + half * 2;          // this lane-half's K pair
            v2f a = *(const v2f*)&As[cur][wv * 16 + l15][ka];
            #pragma unroll
            for (int j = 0; j < 8; ++j) {
                v2f b;
                b.x = Bs[cur][ka + 0][j * 16 + l15];
                b.y = Bs[cur][ka + 1][j * 16 + l15];
                c[j] = __builtin_amdgcn_wmma_f32_16x16x4_f32(
                           false, a, false, b, (short)0, c[j], false, false);
            }
        }
        if (more) cp_async_wait();
        __syncthreads();
    }

    const int rowb = m0 + wv * 16 + half * 8;
    #pragma unroll
    for (int j = 0; j < 8; ++j) {
        int col = j * 16 + l15;
        float bc = bias[col];
        #pragma unroll
        for (int v = 0; v < 8; ++v) {
            float r = c[j][v] + bc;
            if (RELU) r = fmaxf(r, 0.0f);
            out[(size_t)(rowb + v) * HIDD + col] = r;
        }
    }
}

// ---------------------------------------------------------------------------
// h [16384][128] f32  ->  hbT [128][16384] bf16 (transposed, feeds B operand)
// 2 rows per thread -> one packed u32 store.
// ---------------------------------------------------------------------------
__global__ __launch_bounds__(256)
void h_to_bf16T(const float* __restrict__ h, __bf16* __restrict__ hbT)
{
    unsigned idx = blockIdx.x * 256u + threadIdx.x;   // 0 .. 2^20-1 (pairs)
    int n  = idx >> 13;                               // 8192 pairs per column
    int rp = (idx & 8191) * 2;
    float a = h[(size_t)rp * HIDD + n];
    float b = h[(size_t)(rp + 1) * HIDD + n];
    *(unsigned*)(hbT + (size_t)n * NN + rp) = pk_bf16(a, b);
}

// ---------------------------------------------------------------------------
// agg[16384][128] = A @ h + (1+eps)*h   (bf16 WMMA, f32 accum; HBM-bound on A)
// Double-buffered: B tile via async-to-LDS DMA; A tile via pipelined
// regs->cvt_pk_bf16->LDS (conversion forces the register path).
// ---------------------------------------------------------------------------
__global__ __launch_bounds__(128)
void agg_bf16(const float* __restrict__ A, const __bf16* __restrict__ hbT,
              const float* __restrict__ h, const float* __restrict__ eps,
              float* __restrict__ out)
{
    __shared__ __bf16 Asb[2][64][36];   // pitch 36: conflict-free frag reads
    __shared__ __bf16 Bsb[2][128][40];  // pitch 40: 16B-aligned rows

    const int tid  = threadIdx.x;
    const int lane = tid & 31;
    const int wv   = tid >> 5;
    const int m0   = blockIdx.x * 64;
    const int half = lane >> 4;
    const int l15  = lane & 15;

    v8f c[8];
    #pragma unroll
    for (int j = 0; j < 8; ++j) c[j] = (v8f)0.0f;

    float4 areg[4];
    auto loadA = [&](int kb) {
        #pragma unroll
        for (int i = 0; i < 4; ++i) {
            int lin = i * 128 + tid;
            int r = lin >> 3, cc = (lin & 7) * 4;
            const float* src = A + (size_t)(m0 + r) * NN + kb + cc;
            if (((lin & 7) == 0) && (kb + 128 < NN))
                __builtin_prefetch(src + 128, 0, 3);   // global_prefetch_b8
            areg[i] = *(const float4*)src;
        }
    };
    auto storeA = [&](int buf) {
        #pragma unroll
        for (int i = 0; i < 4; ++i) {
            int lin = i * 128 + tid;
            int r = lin >> 3, cc = (lin & 7) * 4;
            uint2 p;
            p.x = pk_bf16(areg[i].x, areg[i].y);
            p.y = pk_bf16(areg[i].z, areg[i].w);
            *(uint2*)&Asb[buf][r][cc] = p;
        }
    };
    auto issueB = [&](int buf, int kb) {
        #pragma unroll
        for (int i = 0; i < 4; ++i) {                  // 4 lanes per 64B row
            int n = i * 32 + (tid >> 2), q = tid & 3;
            cp_async_b128(&Bsb[buf][n][q * 8], hbT + (size_t)n * NN + kb + q * 8);
        }
    };

    issueB(0, 0);
    loadA(0);
    storeA(0);
    cp_async_wait();
    __syncthreads();

    for (int kb = 0; kb < NN; kb += 32) {
        const int cur = (kb >> 5) & 1, nxt = cur ^ 1;
        const bool more = (kb + 32 < NN);
        if (more) { issueB(nxt, kb + 32); loadA(kb + 32); }   // in flight now

        // A-frag: 16-bit A 16x32 layout -> k = (v>>2)*16 + half*8 + (v&3)*2
        AB16 af;
        #pragma unroll
        for (int v = 0; v < 8; ++v) {
            int k = ((v >> 2) * 16) + half * 8 + ((v & 3) * 2);
            af.u[v] = *(const unsigned*)&Asb[cur][wv * 16 + l15][k];
        }
        #pragma unroll
        for (int j = 0; j < 8; ++j) {
            // B-frag: 16-bit B 32x16 layout -> k = (v>>2)*16 + half*8 + (v&3)*2, n = l15
            AB16 bf;
            #pragma unroll
            for (int v = 0; v < 8; ++v) {
                int k = ((v >> 2) * 16) + half * 8 + ((v & 3) * 2);
                bf.u[v] = *(const unsigned*)&Bsb[cur][j * 16 + l15][k];
            }
            c[j] = __builtin_amdgcn_wmma_f32_16x16x32_bf16(
                       false, af.v, false, bf.v, (short)0, c[j], false, false);
        }

        if (more) { storeA(nxt); cp_async_wait(); }    // waits land after WMMAs
        __syncthreads();
    }

    const float scale = 1.0f + eps[0];
    const int rowb = m0 + wv * 16 + half * 8;
    #pragma unroll
    for (int j = 0; j < 8; ++j) {
        int col = j * 16 + l15;
        #pragma unroll
        for (int v = 0; v < 8; ++v) {
            size_t o = (size_t)(rowb + v) * HIDD + col;
            out[o] = c[j][v] + scale * h[o];
        }
    }
}

// ---------------------------------------------------------------------------
// logits = [h|h1|h2] @ W_out + b_out; log_softmax over 16 cols. 1 thread/row.
// ---------------------------------------------------------------------------
__global__ __launch_bounds__(256)
void out_logsoftmax(const float* __restrict__ h, const float* __restrict__ h1,
                    const float* __restrict__ h2, const float* __restrict__ Wo,
                    const float* __restrict__ bo, float* __restrict__ out)
{
    __shared__ float Ws[384 * 16];
    __shared__ float bs[16];
    for (int i = threadIdx.x; i < 384 * 16; i += 256) Ws[i] = Wo[i];
    if (threadIdx.x < 16) bs[threadIdx.x] = bo[threadIdx.x];
    __syncthreads();

    const int r = blockIdx.x * 256 + threadIdx.x;
    float acc[16];
    #pragma unroll
    for (int o = 0; o < 16; ++o) acc[o] = bs[o];

    const float* segs[3] = { h, h1, h2 };
    for (int s = 0; s < 3; ++s) {
        const float* p = segs[s] + (size_t)r * HIDD;
        for (int i = 0; i < HIDD; i += 4) {
            float4 x = *(const float4*)(p + i);
            const float* w = &Ws[(s * HIDD + i) * 16];
            #pragma unroll
            for (int o = 0; o < 16; ++o)
                acc[o] += x.x * w[o] + x.y * w[o + 16] + x.z * w[o + 32] + x.w * w[o + 48];
        }
    }
    float m = acc[0];
    #pragma unroll
    for (int o = 1; o < 16; ++o) m = fmaxf(m, acc[o]);
    float sum = 0.0f;
    #pragma unroll
    for (int o = 0; o < 16; ++o) sum += __expf(acc[o] - m);
    float lse = __logf(sum);
    #pragma unroll
    for (int o = 0; o < 16; ++o) out[(size_t)r * OUTD + o] = acc[o] - m - lse;
}

// ---------------------------------------------------------------------------
extern "C" void kernel_launch(void* const* d_in, const int* in_sizes, int n_in,
                              void* d_out, int out_size, void* d_ws, size_t ws_size,
                              hipStream_t stream)
{
    const float* X    = (const float*)d_in[0];
    const float* A    = (const float*)d_in[1];
    const float* W_in = (const float*)d_in[2];
    const float* b_in = (const float*)d_in[3];
    const float* eps0 = (const float*)d_in[4];
    const float* W1_0 = (const float*)d_in[5];
    const float* b1_0 = (const float*)d_in[6];
    const float* W2_0 = (const float*)d_in[7];
    const float* b2_0 = (const float*)d_in[8];
    const float* eps1 = (const float*)d_in[9];
    const float* W1_1 = (const float*)d_in[10];
    const float* b1_1 = (const float*)d_in[11];
    const float* W2_1 = (const float*)d_in[12];
    const float* b2_1 = (const float*)d_in[13];
    const float* W_o  = (const float*)d_in[14];
    const float* b_o  = (const float*)d_in[15];

    const size_t HN = (size_t)NN * HIDD;
    float*  h   = (float*)d_ws;
    float*  h1  = h  + HN;
    float*  h2  = h1 + HN;
    float*  agg = h2 + HN;
    float*  t   = agg + HN;
    __bf16* hbT = (__bf16*)(t + HN);             // total ws use = 44 MB

    gemm_f32_k<IND,  false><<<256, 128, 0, stream>>>(X,   W_in, b_in, h);

    h_to_bf16T<<<4096, 256, 0, stream>>>(h, hbT);
    agg_bf16<<<256, 128, 0, stream>>>(A, hbT, h, eps0, agg);
    gemm_f32_k<HIDD, true ><<<256, 128, 0, stream>>>(agg, W1_0, b1_0, t);
    gemm_f32_k<HIDD, false><<<256, 128, 0, stream>>>(t,   W2_0, b2_0, h1);

    h_to_bf16T<<<4096, 256, 0, stream>>>(h1, hbT);
    agg_bf16<<<256, 128, 0, stream>>>(A, hbT, h1, eps1, agg);
    gemm_f32_k<HIDD, true ><<<256, 128, 0, stream>>>(agg, W1_1, b1_1, t);
    gemm_f32_k<HIDD, false><<<256, 128, 0, stream>>>(t,   W2_1, b2_1, h2);

    out_logsoftmax<<<64, 256, 0, stream>>>(h, h1, h2, W_o, b_o, (float*)d_out);
}
